// JointCatPoseModel_70222715290002
// MI455X (gfx1250) — matrix operand order repaired and checked by reading, not compiled, not run on hardware
//
#include <hip/hip_runtime.h>

// ---------------- problem constants ----------------
#define BATCH   16384
#define N0      1024
#define CNUM    12
#define KNUM    64
#define DNUM    3
#define NCAT    784              // 16 (y0, padded from 12) + 768 (C*K)
#define MTILE   64
#define BDIM    512              // 16 wave32s
#define NWAVE   16

// ---------------- LDS layout (dynamic) ----------------
#define XS_STRIDE  2064          // bytes/row: 1024 bf16 + 8 pad bf16 (bank-conflict-free)
#define BS_STRIDE  80            // bytes/row: 32 bf16 + 8 pad bf16
#define XS_BYTES   (MTILE * XS_STRIDE)    // 132096
#define BS_BYTES   (NCAT * BS_STRIDE)     // 62720
#define LBB_BYTES  (768 * 4)              // bbin staged as f32
#define IND_BYTES  (768 * 4)              // argmax index per (row, class)
#define SMEM_BYTES (XS_BYTES + BS_BYTES + LBB_BYTES + IND_BYTES)   // 200960

// ---------------- workspace layout (bf16 weights) ----------------
#define WCAT_ELEMS  (NCAT * N0)                 // 802816
#define WREST_ELEMS (CNUM * KNUM * DNUM * N0)   // 2359296
#define WCAT_BYTES  (WCAT_ELEMS * 2)

// ---------------- optional gfx1250 fast paths (probe via __has_builtin) ----------------
#if defined(__has_builtin)
# if __has_builtin(__builtin_amdgcn_global_load_async_to_lds_b128)
#  define HAVE_ASYNC_LDS 1
# else
#  define HAVE_ASYNC_LDS 0
# endif
# if __has_builtin(__builtin_amdgcn_s_wait_asynccnt)
#  define HAVE_WAIT_ASYNC 1
# else
#  define HAVE_WAIT_ASYNC 0
# endif
# if __has_builtin(__builtin_amdgcn_fdot2_f32_bf16)
#  define HAVE_BF16_DOT2 1
# else
#  define HAVE_BF16_DOT2 0
# endif
#else
# define HAVE_ASYNC_LDS 0
# define HAVE_WAIT_ASYNC 0
# define HAVE_BF16_DOT2 0
#endif

#define AS1 __attribute__((address_space(1)))
#define AS3 __attribute__((address_space(3)))

typedef __attribute__((ext_vector_type(16))) __bf16 v16bf;
typedef __attribute__((ext_vector_type(2)))  __bf16 v2bf;
typedef __attribute__((ext_vector_type(8)))  float  v8f;
typedef __attribute__((ext_vector_type(4)))  int    v4i;

union ABReg { v16bf v; uint4 q[2]; };

static __device__ __forceinline__ unsigned short f2bf(float f) {
  unsigned u = __builtin_bit_cast(unsigned, f);
  u += 0x7FFFu + ((u >> 16) & 1u);               // round-to-nearest-even
  return (unsigned short)(u >> 16);
}
static __device__ __forceinline__ float bf2f(unsigned short s) {
  return __builtin_bit_cast(float, ((unsigned)s) << 16);
}
static __device__ __forceinline__ v8f wmma_bf16(v16bf a, v16bf b, v8f c) {
  // D = A(16x32 bf16) * B(32x16 bf16) + C(16x16 f32)
  return __builtin_amdgcn_wmma_f32_16x16x32_bf16(false, a, false, b,
                                                 (short)0, c, false, false);
}
static __device__ __forceinline__ float dot8(uint4 xq, uint4 wq, float acc) {
  unsigned xa[4] = {xq.x, xq.y, xq.z, xq.w};
  unsigned wa[4] = {wq.x, wq.y, wq.z, wq.w};
#pragma unroll
  for (int i = 0; i < 4; ++i) {
#if HAVE_BF16_DOT2
    acc = __builtin_amdgcn_fdot2_f32_bf16(__builtin_bit_cast(v2bf, xa[i]),
                                          __builtin_bit_cast(v2bf, wa[i]),
                                          acc, false);
#else
    acc += bf2f((unsigned short)(xa[i] & 0xFFFFu)) * bf2f((unsigned short)(wa[i] & 0xFFFFu));
    acc += bf2f((unsigned short)(xa[i] >> 16))     * bf2f((unsigned short)(wa[i] >> 16));
#endif
  }
  return acc;
}

// ===== prep: Wcat[784][1024] bf16 = [Wfc^T (12, zero-pad to 16) ; Wbin (768)] =====
__global__ void prep_wcat(const float* __restrict__ Wfc,
                          const float* __restrict__ Wbin,
                          unsigned short* __restrict__ wcat) {
  int idx = blockIdx.x * 256 + threadIdx.x;      // < 802816
  int row = idx >> 10, n = idx & 1023;
  float v;
  if (row < 12)       v = Wfc[n * 12 + row];     // Wfc is [N0, C]
  else if (row < 16)  v = 0.f;
  else                v = Wbin[(row - 16) * N0 + n];
  wcat[idx] = f2bf(v);
}

// ===== prep: WresT[(c*64+k)*3+d][1024] bf16 from Wres[C][K][N0][D] =====
__global__ void prep_wrest(const float* __restrict__ Wres,
                           unsigned short* __restrict__ wrest) {
  int idx = blockIdx.x * 256 + threadIdx.x;      // < 2359296
  int row = idx >> 10, n = idx & 1023;
  int e = row / 3, d = row - e * 3;              // e = c*64+k
  wrest[idx] = f2bf(Wres[((size_t)e * N0 + n) * 3 + d]);
}

// ===== fused: GEMM(bf16 WMMA) + bias + argmax + expert-gather =====
__global__ void __launch_bounds__(BDIM, 1)
joint_catpose_main(const float* __restrict__ x,
                   const float* __restrict__ bfc,
                   const float* __restrict__ bbin_g,
                   const float* __restrict__ bres,
                   const float* __restrict__ centers,
                   const unsigned short* __restrict__ wcat,
                   const unsigned short* __restrict__ wrest,
                   float* __restrict__ out) {
  extern __shared__ char smem[];
  char*     xs  = smem;                                   // x tile, bf16
  char*     bs  = smem + XS_BYTES;                        // B chunk, bf16
  float*    lbb = (float*)(smem + XS_BYTES + BS_BYTES);   // bbin
  unsigned* ind = (unsigned*)(smem + XS_BYTES + BS_BYTES + LBB_BYTES);

  const int tid  = threadIdx.x;
  const int wave = tid >> 5;
  const int lane = tid & 31;
  const int r    = lane & 15;
  const int h    = lane >> 4;
  const int bbase = blockIdx.x * MTILE;

  // ---- Phase A: stage x tile (fp32 -> bf16) into LDS; stage bbin ----
  for (int t = tid; t < (MTILE * N0 / 4); t += BDIM) {    // 16384 float4s
    int row = t >> 8, col4 = (t & 255) << 2;
    float4 f = *(const float4*)(x + (size_t)(bbase + row) * N0 + col4);
    uint2 p;
    p.x = (unsigned)f2bf(f.x) | ((unsigned)f2bf(f.y) << 16);
    p.y = (unsigned)f2bf(f.z) | ((unsigned)f2bf(f.w) << 16);
    *(uint2*)(xs + row * XS_STRIDE + col4 * 2) = p;
  }
  for (int t = tid; t < 768; t += BDIM) lbb[t] = bbin_g[t];

  // ---- wave tile assignment: 48 class-tiles (msub 0..3 x class 0..11) ----
  int msubA[3], cA[3];
#pragma unroll
  for (int i = 0; i < 3; ++i) {
    int id = wave + 16 * i;
    msubA[i] = id / 12;
    cA[i]    = id - msubA[i] * 12;
  }

  v8f accCT[3][4];
  v8f accY0;
#pragma unroll
  for (int i = 0; i < 3; ++i)
#pragma unroll
    for (int j = 0; j < 4; ++j)
#pragma unroll
      for (int e = 0; e < 8; ++e) accCT[i][j][e] = 0.f;
#pragma unroll
  for (int e = 0; e < 8; ++e) accY0[e] = 0.f;

  // ---- Phase B: K-loop, WMMA GEMM ----
  for (int k0 = 0; k0 < N0; k0 += 32) {
    __syncthreads();                       // previous B chunk fully consumed
    // cooperative stage of B chunk [784 rows][32 k] (bf16) from workspace
#if HAVE_ASYNC_LDS
    for (int ch = tid; ch < (NCAT * 4); ch += BDIM) {
      int n = ch >> 2, j = ch & 3;
      AS1 v4i* g = (AS1 v4i*)(unsigned long long)(size_t)(wcat + n * N0 + k0 + j * 8);
      AS3 v4i* l = (AS3 v4i*)(unsigned int)(size_t)(bs + n * BS_STRIDE + j * 16);
      __builtin_amdgcn_global_load_async_to_lds_b128(g, l, 0, 0);
      if (k0 + 32 < N0)                    // warm next chunk into cache
        __builtin_prefetch(wcat + n * N0 + (k0 + 32) + j * 8, 0, 1);
    }
# if HAVE_WAIT_ASYNC
    __builtin_amdgcn_s_wait_asynccnt(0);
# else
    asm volatile("s_wait_asynccnt 0x0" ::: "memory");
# endif
#else
    for (int ch = tid; ch < (NCAT * 4); ch += BDIM) {
      int n = ch >> 2, j = ch & 3;
      uint4 w = *(const uint4*)(wcat + n * N0 + k0 + j * 8);
      *(uint4*)(bs + n * BS_STRIDE + j * 16) = w;
      if (k0 + 32 < N0)                    // warm next chunk into cache
        __builtin_prefetch(wcat + n * N0 + (k0 + 32) + j * 8, 0, 1);
    }
#endif
    __syncthreads();

#pragma unroll
    for (int i = 0; i < 3; ++i) {
      // A operand: row = msub*16 + r, per-lane contiguous K = h*16..h*16+15
      ABReg a;
      const char* ax = xs + (msubA[i] * 16 + r) * XS_STRIDE + (k0 + h * 16) * 2;
      a.q[0] = *(const uint4*)(ax);
      a.q[1] = *(const uint4*)(ax + 16);
      // load ALL four B operands first, then issue 4 back-to-back WMMAs
      ABReg b[4];
#pragma unroll
      for (int j = 0; j < 4; ++j) {
        // B operand: col = r, per-lane contiguous K = h*16..h*16+15
        int n = 16 + cA[i] * 64 + j * 16 + r;
        const char* bp = bs + n * BS_STRIDE + h * 32;
        b[j].q[0] = *(const uint4*)(bp);
        b[j].q[1] = *(const uint4*)(bp + 16);
      }
#pragma unroll
      for (int j = 0; j < 4; ++j)
        accCT[i][j] = wmma_bf16(a.v, b[j].v, accCT[i][j]);
    }
    if (wave < 4) {                        // y0 tile (N-tile 0), msub = wave
      ABReg a;
      const char* ax = xs + (wave * 16 + r) * XS_STRIDE + (k0 + h * 16) * 2;
      a.q[0] = *(const uint4*)(ax);
      a.q[1] = *(const uint4*)(ax + 16);
      const char* bp = bs + r * BS_STRIDE + h * 32;
      ABReg b;
      b.q[0] = *(const uint4*)(bp);
      b.q[1] = *(const uint4*)(bp + 16);
      accY0 = wmma_bf16(a.v, b.v, accY0);
    }
  }

  // ---- y0 output: C/D layout: VGPR v -> row v + 8*h, col r ----
  if (wave < 4 && r < 12) {
    float bias = bfc[r];
#pragma unroll
    for (int v = 0; v < 8; ++v) {
      int row = bbase + wave * 16 + v + 8 * h;
      out[(size_t)row * CNUM + r] = accY0[v] + bias;
    }
  }

  // ---- in-register argmax over K=64 per (row, class) ----
#pragma unroll
  for (int i = 0; i < 3; ++i) {
    const int c = cA[i], msub = msubA[i];
#pragma unroll
    for (int v = 0; v < 8; ++v) {
      float bv = accCT[i][0][v] + lbb[c * 64 + r];
      int   bk = r;
#pragma unroll
      for (int j = 1; j < 4; ++j) {
        float cv = accCT[i][j][v] + lbb[c * 64 + j * 16 + r];
        int   ck = j * 16 + r;
        if (cv > bv || (cv == bv && ck < bk)) { bv = cv; bk = ck; }
      }
      // butterfly over the 16-lane half (rows v / v+8 independent)
#pragma unroll
      for (int off = 8; off >= 1; off >>= 1) {
        float ov = __shfl_xor(bv, off);
        int   ok = __shfl_xor(bk, off);
        if (ov > bv || (ov == bv && ok < bk)) { bv = ov; bk = ok; }
      }
      if (r == 0) ind[(msub * 16 + v + 8 * h) * CNUM + c] = (unsigned)bk;
    }
  }
  __syncthreads();

  // ---- Phase D: gather selected expert, dot against LDS x row ----
  const size_t OUT1 = (size_t)BATCH * CNUM;
  for (int t = wave; t < MTILE * CNUM; t += NWAVE) {
    int bl = t / CNUM, c = t - bl * CNUM;
    int k = (int)ind[bl * CNUM + c];
    // 32 bf16 of x per lane, kept in registers
    const char* xp = xs + bl * XS_STRIDE + lane * 64;
    uint4 xq[4];
#pragma unroll
    for (int q = 0; q < 4; ++q) xq[q] = *(const uint4*)(xp + q * 16);
    int e = c * KNUM + k;
    const unsigned short* wr = wrest + ((size_t)e * 3) * N0 + lane * 32;
    float res[3];
#pragma unroll
    for (int d = 0; d < 3; ++d) {
      const unsigned short* wp = wr + d * N0;
      float acc = 0.f;
#pragma unroll
      for (int q = 0; q < 4; ++q)
        acc = dot8(xq[q], *(const uint4*)(wp + q * 8), acc);
#pragma unroll
      for (int off = 16; off >= 1; off >>= 1) acc += __shfl_xor(acc, off);
      res[d] = acc;
    }
    if (lane == 0) {
      size_t obase = OUT1 + (size_t)(bbase + bl) * (DNUM * CNUM);
#pragma unroll
      for (int d = 0; d < 3; ++d)
        out[obase + d * CNUM + c] = res[d] + bres[e * 3 + d] + centers[k * 3 + d];
    }
  }
}

extern "C" void kernel_launch(void* const* d_in, const int* in_sizes, int n_in,
                              void* d_out, int out_size, void* d_ws, size_t ws_size,
                              hipStream_t stream) {
  const float* x       = (const float*)d_in[0];
  const float* Wfc     = (const float*)d_in[1];
  const float* bfc     = (const float*)d_in[2];
  const float* Wbin    = (const float*)d_in[3];
  const float* bbin    = (const float*)d_in[4];
  const float* Wres    = (const float*)d_in[5];
  const float* bres    = (const float*)d_in[6];
  const float* centers = (const float*)d_in[7];
  float* out = (float*)d_out;

  unsigned short* wcat  = (unsigned short*)d_ws;
  unsigned short* wrest = (unsigned short*)((char*)d_ws + WCAT_BYTES);

  // unconditional (deterministic) — no static guards allowed by the harness
  (void)hipFuncSetAttribute((const void*)joint_catpose_main,
                            hipFuncAttributeMaxDynamicSharedMemorySize,
                            SMEM_BYTES);

  prep_wcat <<<WCAT_ELEMS  / 256, 256, 0, stream>>>(Wfc, Wbin, wcat);
  prep_wrest<<<WREST_ELEMS / 256, 256, 0, stream>>>(Wres, wrest);
  joint_catpose_main<<<BATCH / MTILE, BDIM, SMEM_BYTES, stream>>>(
      x, bfc, bbin, bres, centers, wcat, wrest, out);
}
